// SNNTorchRegressorNet_62517543961117
// MI455X (gfx1250) — matrix operand order repaired and checked by reading, not compiled, not run on hardware
//
#include <hip/hip_runtime.h>
#include <hip/hip_bf16.h>

typedef __attribute__((ext_vector_type(16))) _Float16 v16h;
typedef __attribute__((ext_vector_type(4)))  _Float16 v4h;
typedef __attribute__((ext_vector_type(8)))  float    v8f;
typedef __attribute__((ext_vector_type(4)))  float    v4f;
typedef __attribute__((ext_vector_type(4)))  unsigned int v4u;
typedef __attribute__((ext_vector_type(8)))  int      v8i;
typedef __attribute__((ext_vector_type(4)))  int      v4i;

#define T_STEPS 256
#define BATCH   128
#define HID     1024
#define KT      (HID/32)            /* 32 K-tiles  */
#define NT      (HID/16)            /* 64 N-tiles  */
#define MT      ((T_STEPS*BATCH)/16)/* 2048 M-tiles */
#define FRAG_H  512                 /* halves per 16x16x32 fragment */

#if __has_builtin(__builtin_amdgcn_tensor_load_to_lds)
#define USE_TDM 1
#else
#define USE_TDM 0
#endif

// -------------------------------------------------------------------------
// Kernel 1: layer-1 LIF scan. One thread per (b,h); sequential over T.
// Writes spikes directly in WMMA A-fragment layout (16x32 f16, ISA 7.12.2).
// -------------------------------------------------------------------------
__global__ __launch_bounds__(256) void snn_layer1(
    const float* __restrict__ x, const float* __restrict__ w_in,
    const float* __restrict__ b_in, const float* __restrict__ beta_in,
    const float* __restrict__ thr_in, _Float16* __restrict__ spkA)
{
  int tid = blockIdx.x * 256 + threadIdx.x;     // [0, B*H)
  int b = tid >> 10;
  int h = tid & (HID - 1);
  float bin = fminf(fmaxf(beta_in[h], 0.f), 1.f);
  float thr = thr_in[h];
  float win = w_in[h];
  float bi  = b_in[h];

  int M   = b & 15;
  int kt  = h >> 5;
  int kk  = h & 31;
  int hi  = (kk >> 3) & 1;
  int jj  = (kk & 7) | ((kk >> 4) << 3);
  int lane = M | (hi << 4);
  int mt0  = b >> 4;
  int sub  = lane * 16 + jj;

  float m = 0.f;
  for (int t = 0; t < T_STEPS; ++t) {
    float cur   = x[t * BATCH + b] * win + bi;
    float reset = (m > thr) ? 1.f : 0.f;   // reset from PREVIOUS mem
    m = bin * m + cur - reset * thr;
    _Float16 spk = (m > thr) ? (_Float16)1.0f : (_Float16)0.0f;
    int fidx = ((t * 8 + mt0) * KT + kt) * FRAG_H + sub;
    spkA[fidx] = spk;
  }
}

// -------------------------------------------------------------------------
// Kernel 1b: pack w_h (f32 [n_out][k]) into f16 WMMA B-fragment layout.
// -------------------------------------------------------------------------
__global__ __launch_bounds__(256) void snn_pack_wh(
    const float* __restrict__ wh, _Float16* __restrict__ whB)
{
  int tid = blockIdx.x * 256 + threadIdx.x;     // [0, H*H)
  int n = tid >> 10;          // output column (row of w_h)
  int k = tid & (HID - 1);    // input index
  float v = wh[tid];
  int kt = k >> 5, kk = k & 31;
  int nt = n >> 4, nn = n & 15;
  int hi = (kk >> 3) & 1, jj = (kk & 7) | ((kk >> 4) << 3);
  int lane = nn | (hi << 4);
  int idx = (kt * NT + nt) * FRAG_H + lane * 16 + jj;
  whB[idx] = (_Float16)v;
}

// -------------------------------------------------------------------------
// TDM helper: issue a 1-D tensor DMA of 8 KB (2048 dwords) from global
// memory into LDS at byte offset `ldsoff`.  D# per CDNA5 ISA:
//   group0: count=1, lds_addr, global_addr[56:0], type=2
//   group1: data_size=4B, tensor_dim0=tile_dim0=2048, dim0_stride=2048
// -------------------------------------------------------------------------
#if USE_TDM
__device__ __forceinline__ void tdm_load_8k(unsigned ldsoff, const void* src) {
  unsigned long long ga = (unsigned long long)(uintptr_t)src;
  v4u g0;
  g0[0] = 1u;                                   // count=1, is_restore=0
  g0[1] = ldsoff;                               // lds_addr
  g0[2] = (unsigned)ga;                         // global_addr[31:0]
  g0[3] = (unsigned)((ga >> 32) & 0x01FFFFFFu) | 0x80000000u; // [56:32] | type=2
  v8i g1 = {};
  g1[0] = 0x20000;      // workgroup_mask=0, data_size=2 (4-byte units)
  g1[1] = 0x08000000;   // tensor_dim0 = 2048 (low 16 bits in [31:16])
  g1[2] = 0x00010000;   // tensor_dim1 = 1
  g1[3] = 0x08000000;   // tile_dim0 = 2048  -> 8 KB
  g1[5] = 2048;         // tensor_dim0_stride = 2048
  v4i gz = {};
#if defined(__clang_major__) && (__clang_major__ >= 23)
  v8i gz8 = {};
  __builtin_amdgcn_tensor_load_to_lds(g0, g1, gz, gz, gz8, 0);
#else
  __builtin_amdgcn_tensor_load_to_lds(g0, g1, gz, gz, 0);
#endif
}
#endif

// -------------------------------------------------------------------------
// Kernel 2: GEMM  cur_h = spk1 @ w_h^T  over all T*B rows.
// 8 waves/WG; wave = 16-row M-tile x 128 cols (8 N-tiles).
// B tile (8 KB/K-step) DMA'd into LDS by the Tensor Data Mover (wave 0),
// double-buffered, gated by s_wait_tensorcnt + workgroup barrier.
// A streamed from global with prefetch; accumulators pinned (unroll 1).
// -------------------------------------------------------------------------
__global__ __launch_bounds__(256) void snn_gemm(
    const _Float16* __restrict__ spkA, const _Float16* __restrict__ whB,
    _Float16* __restrict__ curh)
{
  __shared__ v16h Bs[2][256];           // 2 x 8 KB double buffer (LDS offset 0)

  int tid  = threadIdx.x;
  int lane = tid & 31;
  int wave = tid >> 5;
  int mt_g = blockIdx.x * 8 + wave;     // global M-tile
  int ntb  = blockIdx.y * 8;            // first N-tile of this block

  const v16h* Ap = (const v16h*)spkA + mt_g * KT * 32 + lane;  // per-lane A stream
  const v16h* Bg = (const v16h*)whB;                            // B tile stream

  v8f c[8] = {};

  v16h* bufc = &Bs[0][0];
  v16h* bufn = &Bs[1][0];

  // prologue: stage kt=0 B tile into buffer 0
#if USE_TDM
  if (wave == 0) {
    tdm_load_8k(0u, (const void*)(Bg + (0 * NT + ntb) * 32));
    __builtin_amdgcn_s_wait_tensorcnt(0);
  }
#else
  bufc[tid] = Bg[(0 * NT + ntb) * 32 + tid];
#endif
  __syncthreads();

#pragma unroll 1
  for (int kt = 0; kt < KT; ++kt) {
    v16h a = Ap[kt * 32];
    if (kt + 1 < KT) {                                   // stage next tile
#if USE_TDM
      if (wave == 0)
        tdm_load_8k(((unsigned)(kt + 1) & 1u) * 8192u,
                    (const void*)(Bg + ((kt + 1) * NT + ntb) * 32));
#else
      bufn[tid] = Bg[((kt + 1) * NT + ntb) * 32 + tid];
#endif
    }
    if (kt + 4 < KT) {                                   // prefetch A ahead
      __builtin_prefetch((const void*)(Ap + (kt + 4) * 32), 0, 1);
    }

    // pull the 8 B fragments and issue the 8 WMMAs on independent
    // accumulators; ds waits overlap with XDL execution of previous WMMA
    v16h bb[8];
#pragma unroll
    for (int nt = 0; nt < 8; ++nt) bb[nt] = bufc[nt * 32 + lane];
#pragma unroll
    for (int nt = 0; nt < 8; ++nt)
      c[nt] = __builtin_amdgcn_wmma_f32_16x16x32_f16(
          false, a, false, bb[nt], (short)0, c[nt], false, false);

#if USE_TDM
    if (wave == 0) __builtin_amdgcn_s_wait_tensorcnt(0);  // DMA complete
#endif
    __syncthreads();
    v16h* tp = bufc; bufc = bufn; bufn = tp;             // swap buffers
  }

  // C layout (ISA 7.12.2): lane holds (M = r + 8*(lane>>4), N = lane&15)
  int nn = lane & 15, hi = lane >> 4;
  int rowbase = mt_g * 16 + hi * 8;
#pragma unroll
  for (int nt = 0; nt < 8; ++nt) {
    int col = (ntb + nt) * 16 + nn;
#pragma unroll
    for (int r = 0; r < 8; ++r)
      curh[(rowbase + r) * HID + col] = (_Float16)c[nt][r];
  }
}

// -------------------------------------------------------------------------
// Kernel 3: layer-2 leaky scan + spk2 dot w_out + layer-3 leaky scan.
// One workgroup per batch element; deterministic shuffle+LDS reduction.
// -------------------------------------------------------------------------
__global__ __launch_bounds__(256) void snn_out(
    const _Float16* __restrict__ curh, const float* __restrict__ b_h,
    const float* __restrict__ beta_out, const float* __restrict__ w_out,
    const float* __restrict__ b_out, float* __restrict__ out)
{
  __shared__ float red[8];
  int b   = blockIdx.x;
  int tid = threadIdx.x;
  int j0  = tid * 4;
  v4f bh = *(const v4f*)(b_h + j0);
  v4f wo = *(const v4f*)(w_out + j0);
  float bout = fminf(fmaxf(beta_out[0], 0.f), 1.f);
  float bo   = b_out[0];
  float m2[4] = {0.f, 0.f, 0.f, 0.f};
  float m3 = 0.f;
  int lane = tid & 31, wave = tid >> 5;

  for (int t = 0; t < T_STEPS; ++t) {
    v4h ch = *(const v4h*)(curh + (t * BATCH + b) * HID + j0);
    float p = 0.f;
#pragma unroll
    for (int i = 0; i < 4; ++i) {
      float cur = (float)ch[i] + bh[i];
      m2[i] = bout * m2[i] + cur;              // li_out: reset='none'
      p += (m2[i] > 1.0f) ? wo[i] : 0.f;       // spk2 * w_out
    }
#pragma unroll
    for (int msk = 16; msk >= 1; msk >>= 1) p += __shfl_xor(p, msk, 32);
    if (lane == 0) red[wave] = p;
    __syncthreads();
    if (tid == 0) {
      float cur_out = red[0] + red[1] + red[2] + red[3] +
                      red[4] + red[5] + red[6] + red[7] + bo;
      m3 = bout * m3 + cur_out;
      out[t * BATCH + b] = m3;
    }
    __syncthreads();
  }
}

extern "C" void kernel_launch(void* const* d_in, const int* in_sizes, int n_in,
                              void* d_out, int out_size, void* d_ws, size_t ws_size,
                              hipStream_t stream) {
  const float* x        = (const float*)d_in[0];
  const float* w_in     = (const float*)d_in[1];
  const float* b_in     = (const float*)d_in[2];
  const float* beta_in  = (const float*)d_in[3];
  const float* thr_in   = (const float*)d_in[4];
  const float* w_h      = (const float*)d_in[5];
  const float* b_h      = (const float*)d_in[6];
  const float* beta_out = (const float*)d_in[7];
  const float* w_out    = (const float*)d_in[8];
  const float* b_out    = (const float*)d_in[9];
  float* out = (float*)d_out;

  char* ws = (char*)d_ws;
  const size_t spk_bytes = (size_t)T_STEPS * BATCH * HID * sizeof(_Float16); // 64 MB
  const size_t whb_bytes = (size_t)HID * HID * sizeof(_Float16);             // 2 MB
  _Float16* spkA = (_Float16*)ws;
  _Float16* whB  = (_Float16*)(ws + spk_bytes);
  _Float16* curh = (_Float16*)(ws + spk_bytes + whb_bytes);                  // 64 MB

  snn_layer1<<<(BATCH * HID) / 256, 256, 0, stream>>>(x, w_in, b_in, beta_in,
                                                      thr_in, spkA);
  snn_pack_wh<<<(HID * HID) / 256, 256, 0, stream>>>(w_h, whB);
  snn_gemm<<<dim3(MT / 8, HID / 128), 256, 0, stream>>>(spkA, whB, curh);
  snn_out<<<BATCH, 256, 0, stream>>>(curh, b_h, beta_out, w_out, b_out, out);
}
